// MultiHeadAttention_41197326303957
// MI455X (gfx1250) — matrix operand order repaired and checked by reading, not compiled, not run on hardware
//
#include <hip/hip_runtime.h>
#include <math.h>

// ---------------------------------------------------------------------------
// Multi-head attention for MI455X (gfx1250, wave32, WMMA f32_16x16x32_f16)
//
// Roofline: attn_weights output is 536MB fp32 (written by scores, re-read +
// re-written by softmax, re-read by P*V) => ~2.6GB HBM traffic => memory
// bound (~110us @ 23.3TB/s). FLOPs are only ~70G, so all GEMMs go through
// f16 WMMA with f32 accumulate; inputs are ~N(0,1) so f16 is numerically safe.
// ---------------------------------------------------------------------------

typedef _Float16 half_t;
typedef __attribute__((ext_vector_type(16))) _Float16 v16h;
typedef __attribute__((ext_vector_type(8)))  _Float16 v8h;
typedef __attribute__((ext_vector_type(4)))  _Float16 v4h;
typedef __attribute__((ext_vector_type(8)))  float    v8f;
typedef __attribute__((ext_vector_type(4)))  float    v4f;

#define DM   1024   // d_model
#define SEQ  2048   // sequence length
#define NB   2      // batch
#define NH   16     // heads
#define DK   64     // head dim
#define BS   (NB * SEQ)  // 4096 total rows

// ---- WMMA wrapper ---------------------------------------------------------
static __device__ __forceinline__ v8f wmma32(v16h a, v16h b, v8f c) {
  // D(16x16 f32) = A(16x32 f16) * B(32x16 f16) + C
  return __builtin_amdgcn_wmma_f32_16x16x32_f16(
      /*neg_a=*/false, a, /*neg_b=*/false, b,
      /*c_mod=*/(short)0, c, /*reuse_a=*/false, /*reuse_b=*/false);
}

// ---- Fragment loaders (per CDNA5 ISA 7.12.2 per-lane layouts) -------------
// A 16x32 f16: lane l holds row m=l%16; K chunks [hs*8, hs*8+7] and +16.
static __device__ __forceinline__ v16h load_a_f16(const half_t* __restrict__ base, int ld) {
  const int l = threadIdx.x & 31, r = l & 15, hs = l >> 4;
  const half_t* p = base + (size_t)r * ld + hs * 8;
  v8h lo = *(const v8h*)p;
  v8h hi = *(const v8h*)(p + 16);
  v16h a;
#pragma unroll
  for (int i = 0; i < 8; ++i) { a[i] = lo[i]; a[i + 8] = hi[i]; }
  return a;
}

static __device__ __forceinline__ v16h load_a_f32(const float* __restrict__ base, int ld) {
  const int l = threadIdx.x & 31, r = l & 15, hs = l >> 4;
  const float* p = base + (size_t)r * ld + hs * 8;
  v4f f0 = *(const v4f*)p,        f1 = *(const v4f*)(p + 4);
  v4f f2 = *(const v4f*)(p + 16), f3 = *(const v4f*)(p + 20);
  v16h a;
#pragma unroll
  for (int i = 0; i < 4; ++i) {
    a[i]      = (half_t)f0[i];
    a[4 + i]  = (half_t)f1[i];
    a[8 + i]  = (half_t)f2[i];
    a[12 + i] = (half_t)f3[i];
  }
  return a;
}

// B 32x16 f16: lane l holds column n=l%16 (a *row* of the row-major source,
// since we always contract over the contiguous axis); K chunks
// [(l/16)*8, (l/16)*8+7] and +16.
static __device__ __forceinline__ v16h load_b_f16(const half_t* __restrict__ base, int ld) {
  const int l = threadIdx.x & 31, n = l & 15, ko = (l >> 4) * 8;
  const half_t* p = base + (size_t)n * ld + ko;
  v8h lo = *(const v8h*)p;
  v8h hi = *(const v8h*)(p + 16);
  v16h b;
#pragma unroll
  for (int i = 0; i < 8; ++i) { b[i] = lo[i]; b[i + 8] = hi[i]; }
  return b;
}

static __device__ __forceinline__ v16h load_b_f32(const float* __restrict__ base, int ld) {
  const int l = threadIdx.x & 31, n = l & 15, ko = (l >> 4) * 8;
  const float* p = base + (size_t)n * ld + ko;
  v4f f0 = *(const v4f*)p,        f1 = *(const v4f*)(p + 4);
  v4f f2 = *(const v4f*)(p + 16), f3 = *(const v4f*)(p + 20);
  v16h b;
#pragma unroll
  for (int i = 0; i < 4; ++i) {
    b[i]      = (half_t)f0[i];
    b[4 + i]  = (half_t)f1[i];
    b[8 + i]  = (half_t)f2[i];
    b[12 + i] = (half_t)f3[i];
  }
  return b;
}

// ---- fp32 -> fp16 weight conversion ---------------------------------------
__global__ __launch_bounds__(256) void cvt_f32_f16_kernel(
    const float* __restrict__ src, half_t* __restrict__ dst, int n) {
  int i = (blockIdx.x * 256 + threadIdx.x) * 4;
  if (i < n) {
    v4f f = *(const v4f*)(src + i);
    v4h h;
#pragma unroll
    for (int j = 0; j < 4; ++j) h[j] = (half_t)f[j];
    *(v4h*)(dst + i) = h;
  }
}

// ---- Q/K projection: Y[s,f] = sum_k X[s,k]*W[f,k], stored f16 [BS, DM] ----
// Swapped orientation: A=W rows (m=feature), B=X rows (n=seq). D's 8
// contiguous m values per lane => packed v8h row store into Y.
__global__ __launch_bounds__(256) void proj_qk_kernel(
    const half_t* __restrict__ W, const float* __restrict__ X,
    half_t* __restrict__ Y) {
  const int w = threadIdx.x >> 5;
  const int wm = w & 1, wn = w >> 1;
  const int m0 = blockIdx.x * 64 + wm * 32;    // feature
  const int n0 = blockIdx.y * 128 + wn * 32;   // row in [0, BS)
  v8f acc[2][2] = {};
  for (int k0 = 0; k0 < DM; k0 += 32) {
    v16h a0 = load_a_f16(W + (size_t)m0 * DM + k0, DM);
    v16h a1 = load_a_f16(W + (size_t)(m0 + 16) * DM + k0, DM);
    v16h b0 = load_b_f32(X + (size_t)n0 * DM + k0, DM);
    v16h b1 = load_b_f32(X + (size_t)(n0 + 16) * DM + k0, DM);
    acc[0][0] = wmma32(a0, b0, acc[0][0]);
    acc[0][1] = wmma32(a0, b1, acc[0][1]);
    acc[1][0] = wmma32(a1, b0, acc[1][0]);
    acc[1][1] = wmma32(a1, b1, acc[1][1]);
  }
  const int l = threadIdx.x & 31, nl = l & 15, hs = l >> 4;
#pragma unroll
  for (int mi = 0; mi < 2; ++mi)
#pragma unroll
    for (int ni = 0; ni < 2; ++ni) {
      int row = n0 + ni * 16 + nl;
      int f0  = m0 + mi * 16 + hs * 8;
      v8h h;
#pragma unroll
      for (int r = 0; r < 8; ++r) h[r] = (half_t)acc[mi][ni][r];
      *(v8h*)(Y + (size_t)row * DM + f0) = h;
    }
}

// ---- V projection: stores transposed per head: VT[b,h,d,s] f16 ------------
// Normal orientation: A=X rows (m=seq), B=W rows (n=feature). D's contiguous
// m (=seq) per lane => packed v8h store along S of VT.
__global__ __launch_bounds__(256) void proj_v_kernel(
    const float* __restrict__ X, const half_t* __restrict__ W,
    half_t* __restrict__ VT) {
  const int w = threadIdx.x >> 5;
  const int wm = w & 1, wn = w >> 1;
  const int m0 = blockIdx.x * 64 + wm * 32;    // row in [0, BS)
  const int n0 = blockIdx.y * 128 + wn * 32;   // feature
  v8f acc[2][2] = {};
  for (int k0 = 0; k0 < DM; k0 += 32) {
    v16h a0 = load_a_f32(X + (size_t)m0 * DM + k0, DM);
    v16h a1 = load_a_f32(X + (size_t)(m0 + 16) * DM + k0, DM);
    v16h b0 = load_b_f16(W + (size_t)n0 * DM + k0, DM);
    v16h b1 = load_b_f16(W + (size_t)(n0 + 16) * DM + k0, DM);
    acc[0][0] = wmma32(a0, b0, acc[0][0]);
    acc[0][1] = wmma32(a0, b1, acc[0][1]);
    acc[1][0] = wmma32(a1, b0, acc[1][0]);
    acc[1][1] = wmma32(a1, b1, acc[1][1]);
  }
  const int l = threadIdx.x & 31, nl = l & 15, hs = l >> 4;
#pragma unroll
  for (int mi = 0; mi < 2; ++mi)
#pragma unroll
    for (int ni = 0; ni < 2; ++ni) {
      int feat = n0 + ni * 16 + nl;
      int hh = feat >> 6, dd = feat & 63;
      int srow = m0 + mi * 16 + hs * 8;
      int b = srow >> 11, si = srow & (SEQ - 1);
      v8h h;
#pragma unroll
      for (int r = 0; r < 8; ++r) h[r] = (half_t)acc[mi][ni][r];
      *(v8h*)(VT + ((size_t)(b * NH + hh) * DK + dd) * SEQ + si) = h;
    }
}

// ---- Scores: attn[bh,q,kk] = (Qh . Kh)/8, fp32 ----------------------------
// A = Kh rows (m=key), B = Qh rows (n=query). D's contiguous m (=key) per
// lane => coalesced nontemporal v4f stores along each attention row.
__global__ __launch_bounds__(256) void scores_kernel(
    const half_t* __restrict__ KH, const half_t* __restrict__ QH,
    float* __restrict__ attn) {
  const int bh = blockIdx.z;
  const int b = bh >> 4, hh = bh & 15;
  const int col = hh * DK;
  const int w = threadIdx.x >> 5;
  const int wm = w & 1, wn = w >> 1;
  const int m0 = blockIdx.x * 64 + wm * 32;    // key index
  const int n0 = blockIdx.y * 128 + wn * 32;   // query index
  const half_t* Kb = KH + ((size_t)b * SEQ) * DM + col;
  const half_t* Qb = QH + ((size_t)b * SEQ) * DM + col;
  v8f acc[2][2] = {};
#pragma unroll
  for (int k0 = 0; k0 < DK; k0 += 32) {
    v16h a0 = load_a_f16(Kb + (size_t)m0 * DM + k0, DM);
    v16h a1 = load_a_f16(Kb + (size_t)(m0 + 16) * DM + k0, DM);
    v16h b0 = load_b_f16(Qb + (size_t)n0 * DM + k0, DM);
    v16h b1 = load_b_f16(Qb + (size_t)(n0 + 16) * DM + k0, DM);
    acc[0][0] = wmma32(a0, b0, acc[0][0]);
    acc[0][1] = wmma32(a0, b1, acc[0][1]);
    acc[1][0] = wmma32(a1, b0, acc[1][0]);
    acc[1][1] = wmma32(a1, b1, acc[1][1]);
  }
  const float sc = 0.125f;  // 1/sqrt(64)
  const int l = threadIdx.x & 31, nl = l & 15, hs = l >> 4;
#pragma unroll
  for (int mi = 0; mi < 2; ++mi)
#pragma unroll
    for (int ni = 0; ni < 2; ++ni) {
      int q  = n0 + ni * 16 + nl;
      int kk = m0 + mi * 16 + hs * 8;
      float* op = attn + ((size_t)bh * SEQ + q) * SEQ + kk;
      v4f lo = {acc[mi][ni][0] * sc, acc[mi][ni][1] * sc,
                acc[mi][ni][2] * sc, acc[mi][ni][3] * sc};
      v4f hi = {acc[mi][ni][4] * sc, acc[mi][ni][5] * sc,
                acc[mi][ni][6] * sc, acc[mi][ni][7] * sc};
      __builtin_nontemporal_store(lo, (v4f*)op);
      __builtin_nontemporal_store(hi, (v4f*)(op + 4));
    }
}

// ---- Row softmax, in place over attn rows of length SEQ -------------------
__global__ __launch_bounds__(256) void softmax_kernel(float* __restrict__ attn) {
  __shared__ float red[256];
  const int t = threadIdx.x;
  float* p = attn + (size_t)blockIdx.x * SEQ;
  v4f x0 = *(const v4f*)(p + t * 8);
  v4f x1 = *(const v4f*)(p + t * 8 + 4);
  float m = -1e30f;
#pragma unroll
  for (int i = 0; i < 4; ++i) { m = fmaxf(m, x0[i]); m = fmaxf(m, x1[i]); }
  red[t] = m;
  __syncthreads();
  for (int s = 128; s > 0; s >>= 1) {
    if (t < s) red[t] = fmaxf(red[t], red[t + s]);
    __syncthreads();
  }
  m = red[0];
  __syncthreads();
  float e[8], sum = 0.f;
#pragma unroll
  for (int i = 0; i < 4; ++i) {
    e[i]     = __expf(x0[i] - m);
    e[i + 4] = __expf(x1[i] - m);
    sum += e[i] + e[i + 4];
  }
  red[t] = sum;
  __syncthreads();
  for (int s = 128; s > 0; s >>= 1) {
    if (t < s) red[t] += red[t + s];
    __syncthreads();
  }
  float inv = 1.0f / red[0];
  v4f o0 = {e[0] * inv, e[1] * inv, e[2] * inv, e[3] * inv};
  v4f o1 = {e[4] * inv, e[5] * inv, e[6] * inv, e[7] * inv};
  __builtin_nontemporal_store(o0, (v4f*)(p + t * 8));
  __builtin_nontemporal_store(o1, (v4f*)(p + t * 8 + 4));
}

// ---- P*V: AO[b,s, h*64+d] f16 = sum_k P[q,k] * VT[d,k] --------------------
// A = VT rows (m=d, contiguous over keys), B = P rows (f32->f16, n=query).
// D's contiguous m (=d) per lane => packed v8h store into head-concat layout.
__global__ __launch_bounds__(256) void pv_kernel(
    const half_t* __restrict__ VT, const float* __restrict__ attn,
    half_t* __restrict__ AO) {
  const int bh = blockIdx.z;
  const int b = bh >> 4, hh = bh & 15;
  const int w = threadIdx.x >> 5;
  const int n0 = blockIdx.x * 256 + w * 32;    // query index
  const half_t* Ab = VT + (size_t)bh * DK * SEQ;
  const float* Bb = attn + (size_t)bh * SEQ * SEQ;
  v8f acc[4][2] = {};
  for (int k0 = 0; k0 < SEQ; k0 += 32) {
    if (k0 + 64 < SEQ)  // prefetch next P tile (streams from HBM)
      __builtin_prefetch(Bb + (size_t)n0 * SEQ + k0 + 64, 0, 1);
    v16h a[4], bb[2];
#pragma unroll
    for (int mi = 0; mi < 4; ++mi)
      a[mi] = load_a_f16(Ab + (size_t)(mi * 16) * SEQ + k0, SEQ);
#pragma unroll
    for (int ni = 0; ni < 2; ++ni)
      bb[ni] = load_b_f32(Bb + (size_t)(n0 + ni * 16) * SEQ + k0, SEQ);
#pragma unroll
    for (int mi = 0; mi < 4; ++mi)
#pragma unroll
      for (int ni = 0; ni < 2; ++ni)
        acc[mi][ni] = wmma32(a[mi], bb[ni], acc[mi][ni]);
  }
  const int l = threadIdx.x & 31, nl = l & 15, hs = l >> 4;
#pragma unroll
  for (int mi = 0; mi < 4; ++mi)
#pragma unroll
    for (int ni = 0; ni < 2; ++ni) {
      int q  = n0 + ni * 16 + nl;
      int d0 = mi * 16 + hs * 8;
      v8h h;
#pragma unroll
      for (int r = 0; r < 8; ++r) h[r] = (half_t)acc[mi][ni][r];
      *(v8h*)(AO + (size_t)(b * SEQ + q) * DM + hh * DK + d0) = h;
    }
}

// ---- Output projection: OUT[s,f] f32 = sum_k AO[s,k]*Wo[f,k] --------------
__global__ __launch_bounds__(256) void outproj_kernel(
    const half_t* __restrict__ W, const half_t* __restrict__ X,
    float* __restrict__ OUT) {
  const int w = threadIdx.x >> 5;
  const int wm = w & 1, wn = w >> 1;
  const int m0 = blockIdx.x * 64 + wm * 32;    // output feature
  const int n0 = blockIdx.y * 128 + wn * 32;   // row in [0, BS)
  v8f acc[2][2] = {};
  for (int k0 = 0; k0 < DM; k0 += 32) {
    v16h a0 = load_a_f16(W + (size_t)m0 * DM + k0, DM);
    v16h a1 = load_a_f16(W + (size_t)(m0 + 16) * DM + k0, DM);
    v16h b0 = load_b_f16(X + (size_t)n0 * DM + k0, DM);
    v16h b1 = load_b_f16(X + (size_t)(n0 + 16) * DM + k0, DM);
    acc[0][0] = wmma32(a0, b0, acc[0][0]);
    acc[0][1] = wmma32(a0, b1, acc[0][1]);
    acc[1][0] = wmma32(a1, b0, acc[1][0]);
    acc[1][1] = wmma32(a1, b1, acc[1][1]);
  }
  const int l = threadIdx.x & 31, nl = l & 15, hs = l >> 4;
#pragma unroll
  for (int mi = 0; mi < 2; ++mi)
#pragma unroll
    for (int ni = 0; ni < 2; ++ni) {
      int row = n0 + ni * 16 + nl;
      int f0  = m0 + mi * 16 + hs * 8;
      float* op = OUT + (size_t)row * DM + f0;
      v4f lo = {acc[mi][ni][0], acc[mi][ni][1], acc[mi][ni][2], acc[mi][ni][3]};
      v4f hi = {acc[mi][ni][4], acc[mi][ni][5], acc[mi][ni][6], acc[mi][ni][7]};
      *(v4f*)op = lo;
      *(v4f*)(op + 4) = hi;
    }
}

// ---------------------------------------------------------------------------
extern "C" void kernel_launch(void* const* d_in, const int* in_sizes, int n_in,
                              void* d_out, int out_size, void* d_ws, size_t ws_size,
                              hipStream_t stream) {
  (void)in_sizes; (void)n_in; (void)out_size; (void)ws_size;
  const float* q  = (const float*)d_in[0];
  const float* k  = (const float*)d_in[1];
  const float* v  = (const float*)d_in[2];
  const float* Wq = (const float*)d_in[3];
  const float* Wk = (const float*)d_in[4];
  const float* Wv = (const float*)d_in[5];
  const float* Wo = (const float*)d_in[6];

  float* out  = (float*)d_out;                        // [BS, DM] fp32
  float* attn = out + (size_t)BS * DM;                // [NB*NH, SEQ, SEQ] fp32

  // Workspace layout (f16 units): 4x weight (1M each) + QH/KH (4M each) +
  // VT (4M) + AO (4M) = 20M halves = 40 MB.
  const size_t M1 = (size_t)DM * DM;       // 1M
  const size_t M4 = (size_t)BS * DM;       // 4M
  half_t* wsh  = (half_t*)d_ws;
  half_t* Wq16 = wsh;
  half_t* Wk16 = wsh + M1;
  half_t* Wv16 = wsh + 2 * M1;
  half_t* Wo16 = wsh + 3 * M1;
  half_t* QH   = wsh + 4 * M1;             // [BS, DM]
  half_t* KH   = QH + M4;
  half_t* VT   = KH + M4;                  // [NB*NH, DK, SEQ]
  half_t* AO   = VT + M4;                  // [BS, DM]

  // 1) fp32 -> fp16 weights
  cvt_f32_f16_kernel<<<1024, 256, 0, stream>>>(Wq, Wq16, (int)M1);
  cvt_f32_f16_kernel<<<1024, 256, 0, stream>>>(Wk, Wk16, (int)M1);
  cvt_f32_f16_kernel<<<1024, 256, 0, stream>>>(Wv, Wv16, (int)M1);
  cvt_f32_f16_kernel<<<1024, 256, 0, stream>>>(Wo, Wo16, (int)M1);

  // 2) projections
  proj_qk_kernel<<<dim3(DM / 64, BS / 128), 256, 0, stream>>>(Wq16, q, QH);
  proj_qk_kernel<<<dim3(DM / 64, BS / 128), 256, 0, stream>>>(Wk16, k, KH);
  proj_v_kernel<<<dim3(BS / 64, DM / 128), 256, 0, stream>>>(v, Wv16, VT);

  // 3) scores = Q.K^T / sqrt(dk)  -> attn region (raw, fp32)
  scores_kernel<<<dim3(SEQ / 64, SEQ / 128, NB * NH), 256, 0, stream>>>(KH, QH, attn);

  // 4) softmax in place over each attention row
  softmax_kernel<<<NB * NH * SEQ, 256, 0, stream>>>(attn);

  // 5) P*V -> head-concat f16 activations
  pv_kernel<<<dim3(SEQ / 256, 1, NB * NH), 256, 0, stream>>>(VT, attn, AO);

  // 6) output projection -> fp32 output region
  outproj_kernel<<<dim3(DM / 64, BS / 128), 256, 0, stream>>>(Wo16, AO, out);
}
